// AttentionHead_33294586478609
// MI455X (gfx1250) — matrix operand order repaired and checked by reading, not compiled, not run on hardware
//
#include <hip/hip_runtime.h>
#include <hip/hip_bf16.h>
#include <stdint.h>

typedef __attribute__((ext_vector_type(16))) _Float16 v16h;
typedef __attribute__((ext_vector_type(8)))  float    v8f;
typedef __attribute__((ext_vector_type(4)))  uint32_t u32x4;
typedef __attribute__((ext_vector_type(4)))  int      i32x4;
typedef __attribute__((ext_vector_type(8)))  int      i32x8;

#define BATCH 32
#define CHN   256
#define SEQ   1024
// (1/sqrt(256)) * log2(e): fold softmax base-2 conversion into Q scaling
#define QSCALE 0.09016844005556021f

#if defined(__has_builtin)
#  if __has_builtin(__builtin_amdgcn_tensor_load_to_lds)
#    define HAVE_TDM 1
#  endif
#endif
#ifndef HAVE_TDM
#  define HAVE_TDM 0
#endif

__device__ __forceinline__ uint32_t lds_addr_of(const void* p) {
    // LDS-aperture flat address: low 32 bits are the LDS byte offset
    return (uint32_t)(uintptr_t)p;
}

__device__ __forceinline__ void wait_tensorcnt0() {
#if defined(__has_builtin) && __has_builtin(__builtin_amdgcn_s_wait_tensorcnt)
    __builtin_amdgcn_s_wait_tensorcnt(0);
#else
    asm volatile("s_wait_tensorcnt 0x0" ::: "memory");
#endif
}

#if HAVE_TDM
// 2-D TDM load: tile (tile_d1 rows x tile_d0 elems) of a (tensor_d1 x tensor_d0)
// f16 tensor with row stride d0_stride, into LDS at lds_off (row-major packed).
__device__ __forceinline__ void tdm_load_2d(uint32_t lds_off, const void* gptr,
        uint32_t tensor_d0, uint32_t tensor_d1,
        uint32_t tile_d0, uint32_t tile_d1, uint32_t d0_stride) {
    uint64_t ga = (uint64_t)(uintptr_t)gptr;
    u32x4 g0 = { 1u,                                    // count=1 (valid user D#)
                 lds_off,                               // lds_addr
                 (uint32_t)ga,                          // global_addr[31:0]
                 (uint32_t)(ga >> 32) | (2u << 30) };   // global_addr[56:32] | type=2
    i32x8 g1 = { (int)(1u << 16),                       // data_size=1 (2 bytes)
                 (int)((tensor_d0 & 0xFFFFu) << 16),    // tensor_dim0[15:0]
                 (int)((tensor_d0 >> 16) | ((tensor_d1 & 0xFFFFu) << 16)),
                 (int)((tensor_d1 >> 16) | (tile_d0 << 16)),
                 (int)(tile_d1 & 0xFFFFu),              // tile_dim1 (tile_dim2=0)
                 (int)d0_stride,                        // tensor_dim0_stride[31:0]
                 0, 0 };
    i32x4 z4 = {0, 0, 0, 0};
#if __clang_major__ >= 23
    i32x8 z8 = {0, 0, 0, 0, 0, 0, 0, 0};
    __builtin_amdgcn_tensor_load_to_lds(g0, g1, z4, z4, z8, 0);
#else
    __builtin_amdgcn_tensor_load_to_lds(g0, g1, z4, z4, 0);
#endif
}
#endif

// A-fragment (16x32 f16) K-pair base for vgpr v, half h=(lane>>4)
__device__ __forceinline__ int kpb(int v, int h) {
    return (v < 4) ? (2 * v + 8 * h) : (16 + 2 * (v - 4) + 8 * h);
}

// ---------------- kernel 0a: x [B,C,S] f32 -> xh [B,S,C] f16 ----------------
__global__ void transpose_x_kernel(const float* __restrict__ x, _Float16* __restrict__ xh) {
    __shared__ float tile[32][33];
    int b  = blockIdx.z;
    int c0 = blockIdx.y * 32;
    int s0 = blockIdx.x * 32;
    int tx = threadIdx.x;   // 0..31
    int ty = threadIdx.y;   // 0..7
    const float* xp = x + (size_t)b * CHN * SEQ;
    #pragma unroll
    for (int i = 0; i < 32; i += 8)
        tile[ty + i][tx] = xp[(size_t)(c0 + ty + i) * SEQ + s0 + tx];
    __syncthreads();
    _Float16* xo = xh + (size_t)b * SEQ * CHN;
    #pragma unroll
    for (int i = 0; i < 32; i += 8)
        xo[(size_t)(s0 + ty + i) * CHN + c0 + tx] = (_Float16)tile[tx][ty + i];
}

// ---------------- kernel 0b: W [d,c] f32 -> Wt [c,d] f16 ----------------
__global__ void transpose_w_kernel(const float* __restrict__ w, _Float16* __restrict__ wt) {
    int idx = blockIdx.x * blockDim.x + threadIdx.x;   // 0..65535
    int c = idx >> 8, d = idx & 255;
    wt[idx] = (_Float16)w[d * 256 + c];
}

// ---------------- kernel 1: QKV projections via WMMA ----------------
// grid (B*S/64), 128 threads (4 waves, 1 per SIMD -> full VGPR file).
// Block: 64 rows x all 256 cols x {Q,K,V}. x tile staged in LDS via TDM.
__global__ void __launch_bounds__(128, 1)
qkv_kernel(const _Float16* __restrict__ xh,
        const _Float16* __restrict__ wqt, const _Float16* __restrict__ wkt,
        const _Float16* __restrict__ wvt,
        const float* __restrict__ bq, const float* __restrict__ bk,
        const float* __restrict__ bv,
        _Float16* __restrict__ qh, _Float16* __restrict__ ktr,
        _Float16* __restrict__ vh) {
    __shared__ _Float16 xt[64][CHN];   // 32 KB
    int tid  = threadIdx.x;
    int lane = tid & 31;
    int wv_  = tid >> 5;           // wave 0..3
    int l15  = lane & 15;
    int h    = lane >> 4;

    int b    = blockIdx.x >> 4;            // 16 row-tiles per batch
    int srow = (blockIdx.x & 15) << 6;     // 64 rows

    // stage x tile [64 x 256] f16 (contiguous in xh)
#if HAVE_TDM
    if (wv_ == 0) {
        tdm_load_2d(lds_addr_of(&xt[0][0]), xh + (size_t)(b * SEQ + srow) * CHN,
                    /*tensor_d0=*/CHN, /*tensor_d1=*/64,
                    /*tile_d0=*/CHN, /*tile_d1=*/64, /*d0_stride=*/CHN);
        wait_tensorcnt0();
    }
    __syncthreads();
#else
    {
        const uint32_t* src = (const uint32_t*)(xh + (size_t)(b * SEQ + srow) * CHN);
        uint32_t* dst = (uint32_t*)&xt[0][0];
        for (int i = tid; i < 64 * CHN / 2; i += 128) dst[i] = src[i];
        __syncthreads();
    }
#endif

    const _Float16* wts[3]    = { wqt, wkt, wvt };
    const float*    biases[3] = { bq, bk, bv };

    for (int which = 0; which < 3; ++which) {
        const _Float16* wt   = wts[which];
        const float*    bias = biases[which];

        v8f acc[4][4] = {};   // [mt][nt]
        for (int ck = 0; ck < 8; ++ck) {
            int c0 = ck * 32;
            union { v16h hh; uint32_t u[8]; } af[4];
            #pragma unroll
            for (int mt = 0; mt < 4; ++mt)
                #pragma unroll
                for (int v = 0; v < 8; ++v)
                    af[mt].u[v] = *(const uint32_t*)&xt[mt * 16 + l15][c0 + kpb(v, h)];
            #pragma unroll
            for (int nt = 0; nt < 4; ++nt) {
                int n0 = wv_ * 64 + nt * 16;
                v16h bf;
                #pragma unroll
                for (int s = 0; s < 16; ++s)
                    bf[s] = wt[(size_t)(c0 + 8 * h + s + ((s >> 3) << 3)) * CHN + n0 + l15];
                #pragma unroll
                for (int mt = 0; mt < 4; ++mt)
                    acc[mt][nt] = __builtin_amdgcn_wmma_f32_16x16x32_f16(
                        false, af[mt].hh, false, bf, (short)0, acc[mt][nt], false, false);
            }
        }
        #pragma unroll
        for (int nt = 0; nt < 4; ++nt) {
            int ncol = wv_ * 64 + nt * 16 + l15;
            float bb = bias[ncol];
            #pragma unroll
            for (int mt = 0; mt < 4; ++mt) {
                int rowb = srow + mt * 16 + 8 * h;
                if (which == 1) {
                    _Float16* kp = ktr + (size_t)b * CHN * SEQ + (size_t)ncol * SEQ + rowb;
                    #pragma unroll
                    for (int r = 0; r < 8; ++r)
                        kp[r] = (_Float16)(acc[mt][nt][r] + bb);
                } else {
                    _Float16* op = (which == 0 ? qh : vh)
                                 + (size_t)(b * SEQ + rowb) * CHN + ncol;
                    float s = (which == 0) ? QSCALE : 1.0f;
                    #pragma unroll
                    for (int r = 0; r < 8; ++r)
                        op[(size_t)r * CHN] = (_Float16)((acc[mt][nt][r] + bb) * s);
                }
            }
        }
    }
}

// ---------------- kernel 2: flash attention + SiLU + transpose-out ----------------
// 256 threads (8 waves, 2 per SIMD); wave owns 16 query rows; block = 128 rows.
// K/V tiles double-buffered in LDS via TDM, shared by all 8 waves.
__global__ void __launch_bounds__(256, 1)
attn_kernel(const _Float16* __restrict__ qh,
            const _Float16* __restrict__ ktr,
            const _Float16* __restrict__ vh,
            float* __restrict__ out) {
    __shared__ _Float16 ktile[2][CHN][32];   // 2 x 16 KB  (Kt rows c, cols key)
    __shared__ _Float16 vtile[2][32][CHN];   // 2 x 16 KB  (rows key, cols d)
    __shared__ _Float16 lds_p[8][16][32];    // 8 KB per-wave P staging
    int tid  = threadIdx.x;
    int lane = tid & 31;
    int wv_  = tid >> 5;
    int l15  = lane & 15;
    int h    = lane >> 4;

    int bpb  = SEQ / 128;
    int b    = blockIdx.x / bpb;
    int srow = (blockIdx.x % bpb) * 128 + wv_ * 16;

    const _Float16* qrow  = qh  + (size_t)(b * SEQ + srow + l15) * CHN;
    const _Float16* kbase = ktr + (size_t)b * CHN * SEQ;
    const _Float16* vbase = vh  + (size_t)b * SEQ * CHN;

    union AF { v16h hh; uint32_t u[8]; };
    AF qa[8];                                 // resident Q fragments (16x256)
    #pragma unroll
    for (int ck = 0; ck < 8; ++ck)
        #pragma unroll
        for (int v = 0; v < 8; ++v)
            qa[ck].u[v] = *(const uint32_t*)(qrow + ck * 32 + kpb(v, h));

    v8f o[16] = {};
    float mi[8], li[8];
    #pragma unroll
    for (int r = 0; r < 8; ++r) { mi[r] = -3.0e38f; li[r] = 0.0f; }

    const int NKT = SEQ / 32;

    // ---- stage tile 0 into buffer 0 ----
#if HAVE_TDM
    if (wv_ == 0) {
        tdm_load_2d(lds_addr_of(&ktile[0][0][0]), kbase, SEQ, CHN, 32, CHN, SEQ);
        tdm_load_2d(lds_addr_of(&vtile[0][0][0]), vbase, CHN, SEQ, CHN, 32, CHN);
        wait_tensorcnt0();
    }
    __syncthreads();
#else
    {
        uint32_t* kd = (uint32_t*)&ktile[0][0][0];
        uint32_t* vd = (uint32_t*)&vtile[0][0][0];
        for (int i = tid; i < CHN * 16; i += 256) {
            int row = i >> 4, seg = i & 15;
            kd[i] = *(const uint32_t*)(kbase + (size_t)row * SEQ + seg * 2);
        }
        for (int i = tid; i < 32 * CHN / 2; i += 256) {
            int row = i >> 7, seg = i & 127;
            vd[i] = *(const uint32_t*)(vbase + (size_t)row * CHN + seg * 2);
        }
        __syncthreads();
    }
#endif

    for (int kt = 0; kt < NKT; ++kt) {
        int cur = kt & 1;
        // prefetch next tile into the other buffer (overlaps with consume)
        if (kt + 1 < NKT) {
            int key1 = (kt + 1) * 32;
#if HAVE_TDM
            if (wv_ == 0) {
                tdm_load_2d(lds_addr_of(&ktile[cur ^ 1][0][0]), kbase + key1,
                            SEQ, CHN, 32, CHN, SEQ);
                tdm_load_2d(lds_addr_of(&vtile[cur ^ 1][0][0]),
                            vbase + (size_t)key1 * CHN, CHN, SEQ, CHN, 32, CHN);
            }
#else
            uint32_t* kd = (uint32_t*)&ktile[cur ^ 1][0][0];
            uint32_t* vd = (uint32_t*)&vtile[cur ^ 1][0][0];
            for (int i = tid; i < CHN * 16; i += 256) {
                int row = i >> 4, seg = i & 15;
                kd[i] = *(const uint32_t*)(kbase + (size_t)row * SEQ + key1 + seg * 2);
            }
            for (int i = tid; i < 32 * CHN / 2; i += 256) {
                int row = i >> 7, seg = i & 127;
                vd[i] = *(const uint32_t*)(vbase + (size_t)(key1 + row) * CHN + seg * 2);
            }
#endif
        }

        // ---- scores: S = Q x K^T (from LDS) ----
        v8f sc0 = {}, sc1 = {};
        #pragma unroll
        for (int ck = 0; ck < 8; ++ck) {
            v16h b0, b1;
            #pragma unroll
            for (int s = 0; s < 16; ++s) {
                const _Float16* kr = &ktile[cur][ck * 32 + 8 * h + s + ((s >> 3) << 3)][0];
                b0[s] = kr[l15];
                b1[s] = kr[l15 + 16];
            }
            sc0 = __builtin_amdgcn_wmma_f32_16x16x32_f16(
                false, qa[ck].hh, false, b0, (short)0, sc0, false, false);
            sc1 = __builtin_amdgcn_wmma_f32_16x16x32_f16(
                false, qa[ck].hh, false, b1, (short)0, sc1, false, false);
        }
        // ---- online softmax (base-2 domain; log2(e) folded into Q scale) ----
        #pragma unroll
        for (int r = 0; r < 8; ++r) {
            float a0 = sc0[r], a1 = sc1[r];
            float mx = fmaxf(a0, a1);
            #pragma unroll
            for (int msk = 1; msk < 16; msk <<= 1)
                mx = fmaxf(mx, __shfl_xor(mx, msk, 32));
            float mnew  = fmaxf(mi[r], mx);
            float alpha = exp2f(mi[r] - mnew);
            float p0 = exp2f(a0 - mnew);
            float p1 = exp2f(a1 - mnew);
            float rs = p0 + p1;
            #pragma unroll
            for (int msk = 1; msk < 16; msk <<= 1)
                rs += __shfl_xor(rs, msk, 32);
            li[r] = li[r] * alpha + rs;
            mi[r] = mnew;
            #pragma unroll
            for (int nt = 0; nt < 16; ++nt)
                o[nt][r] *= alpha;
            int m = r + 8 * h;
            lds_p[wv_][m][l15]      = (_Float16)p0;
            lds_p[wv_][m][l15 + 16] = (_Float16)p1;
        }
        // P: C-layout -> A-layout via per-wave LDS (same-wave DS ops in order)
        AF pf;
        #pragma unroll
        for (int v = 0; v < 8; ++v)
            pf.u[v] = *(const uint32_t*)&lds_p[wv_][l15][kpb(v, h)];
        // ---- O += P x V (from LDS) ----
        #pragma unroll
        for (int nt = 0; nt < 16; ++nt) {
            v16h bvf;
            #pragma unroll
            for (int s = 0; s < 16; ++s)
                bvf[s] = vtile[cur][8 * h + s + ((s >> 3) << 3)][nt * 16 + l15];
            o[nt] = __builtin_amdgcn_wmma_f32_16x16x32_f16(
                false, pf.hh, false, bvf, (short)0, o[nt], false, false);
        }

        __syncthreads();               // all waves done consuming buffer `cur`
#if HAVE_TDM
        if (wv_ == 0) wait_tensorcnt0();   // next buffer fully landed
#endif
        __syncthreads();
    }

    // ---- finalize: 1/l, SiLU, store transposed [B,C,S] ----
    float inv[8];
    #pragma unroll
    for (int r = 0; r < 8; ++r) inv[r] = 1.0f / li[r];
    float* obase = out + (size_t)b * CHN * SEQ;
    #pragma unroll
    for (int nt = 0; nt < 16; ++nt) {
        int ncol = nt * 16 + l15;
        float* op = obase + (size_t)ncol * SEQ + srow + 8 * h;
        #pragma unroll
        for (int r = 0; r < 8; ++r) {
            float v = o[nt][r] * inv[r];
            v = v / (1.0f + __expf(-v));   // SiLU
            op[r] = v;
        }
    }
}

extern "C" void kernel_launch(void* const* d_in, const int* in_sizes, int n_in,
                              void* d_out, int out_size, void* d_ws, size_t ws_size,
                              hipStream_t stream) {
    (void)in_sizes; (void)n_in; (void)out_size; (void)ws_size;
    const float* x  = (const float*)d_in[0];
    const float* wq = (const float*)d_in[1];
    const float* bq = (const float*)d_in[2];
    const float* wk = (const float*)d_in[3];
    const float* bk = (const float*)d_in[4];
    const float* wv = (const float*)d_in[5];
    const float* bv = (const float*)d_in[6];
    float* out = (float*)d_out;

    _Float16* ws = (_Float16*)d_ws;
    const size_t NSC = (size_t)BATCH * SEQ * CHN;   // 8,388,608 halfs per tensor
    _Float16* xh  = ws;
    _Float16* qh  = ws + 1 * NSC;
    _Float16* ktr = ws + 2 * NSC;
    _Float16* vh  = ws + 3 * NSC;
    _Float16* wqt = ws + 4 * NSC;
    _Float16* wkt = wqt + 65536;
    _Float16* wvt = wkt + 65536;

    transpose_x_kernel<<<dim3(SEQ / 32, CHN / 32, BATCH), dim3(32, 8), 0, stream>>>(x, xh);
    transpose_w_kernel<<<256, 256, 0, stream>>>(wq, wqt);
    transpose_w_kernel<<<256, 256, 0, stream>>>(wk, wkt);
    transpose_w_kernel<<<256, 256, 0, stream>>>(wv, wvt);
    qkv_kernel<<<dim3(BATCH * SEQ / 64), 128, 0, stream>>>(
        xh, wqt, wkt, wvt, bq, bk, bv, qh, ktr, vh);
    attn_kernel<<<dim3(BATCH * SEQ / 128), 256, 0, stream>>>(qh, ktr, vh, out);
}
